// MultiheadAttention_10617159155992
// MI455X (gfx1250) — matrix operand order repaired and checked by reading, not compiled, not run on hardware
//
#include <hip/hip_runtime.h>
#include <hip/hip_bf16.h>

typedef __bf16 bf16_t;
typedef __attribute__((ext_vector_type(16))) __bf16 v16bf;
typedef __attribute__((ext_vector_type(8)))  __bf16 v8bf;
typedef __attribute__((ext_vector_type(8)))  float  v8f;

#define L_SEQ 2048
#define N_B   2
#define E_DIM 1024
#define H_N   16
#define T_N   128
#define DH    64

static __device__ inline v8f wmma_bf16(v16bf a, v16bf b, v8f c) {
  // (neg_a, A, neg_b, B, c_mod, C, reuse_a, reuse_b)
  return __builtin_amdgcn_wmma_f32_16x16x32_bf16(false, a, false, b, (short)0, c,
                                                 false, false);
}

// ---- Operand loaders (wave32 WMMA, contiguous k order shared by A and B) ----
// A operand: 16x32 (MxK) bf16 tile. Lane m=lane&15; lanes<16 hold K 0-15,
// lanes>=16 hold K 16-31 (v16bf elems 0-15).
static __device__ inline v16bf load_a_bf16(const bf16_t* src, int row0, int k0,
                                           int ld, int lane) {
  int m = lane & 15, hi = lane >> 4;
  const bf16_t* p = src + (size_t)(row0 + m) * ld + k0 + hi * 16;
  v8bf lo = *(const v8bf*)p;
  v8bf hi8 = *(const v8bf*)(p + 8);
  v16bf a;
#pragma unroll
  for (int i = 0; i < 8; ++i) { a[i] = lo[i]; a[8 + i] = hi8[i]; }
  return a;
}

// B operand: 32x16 (KxN) tile, B[k][n] = src[n0+n][k0+k] (src row-major NxK).
// Lane n=lane&15; lanes<16 hold K 0-15, lanes>=16 hold K 16-31.
static __device__ inline v16bf load_b_bf16(const bf16_t* src, int n0, int k0,
                                           int ld, int lane) {
  int n = lane & 15, hi = lane >> 4;
  const bf16_t* p = src + (size_t)(n0 + n) * ld + k0 + hi * 16;
  v8bf lo = *(const v8bf*)p;
  v8bf hi8 = *(const v8bf*)(p + 8);
  v16bf b;
#pragma unroll
  for (int i = 0; i < 8; ++i) { b[i] = lo[i]; b[8 + i] = hi8[i]; }
  return b;
}

// ---- Kernel 0: one-shot fp32 -> bf16 conversion (8 elems/thread) ----
__global__ __launch_bounds__(256) void cvt_bf16_kernel(
    const float* __restrict__ src, bf16_t* __restrict__ dst, int n8) {
  int i = blockIdx.x * 256 + threadIdx.x;
  if (i >= n8) return;
  v8f x = *(const v8f*)(src + (size_t)i * 8);
  v8bf y;
#pragma unroll
  for (int j = 0; j < 8; ++j) y[j] = (__bf16)x[j];
  *(v8bf*)(dst + (size_t)i * 8) = y;
}

// ---- Kernel 1: fused QKV projection, 16x64 tile per wave ----
// grid: ((L*N)/16, 3E/64), block 32. Pure bf16 loads + WMMA inner loop.
__global__ __launch_bounds__(32) void qkv_proj_kernel(
    const bf16_t* __restrict__ Xq, const bf16_t* __restrict__ Xk,
    const bf16_t* __restrict__ Xv, const bf16_t* __restrict__ Wbf,
    const float* __restrict__ ipb, bf16_t* __restrict__ Qo,
    bf16_t* __restrict__ Ko, bf16_t* __restrict__ Vto) {
  int lane = threadIdx.x;
  int mt = blockIdx.x * 16;       // row tile over L*N
  int jt = blockIdx.y * 64;       // col tile within 3E (never straddles E)
  int which = jt >> 10;           // jt / E_DIM: 0=q 1=k 2=v
  const bf16_t* X = (which == 0) ? Xq : (which == 1) ? Xk : Xv;
  v8f a0 = {}, a1 = {}, a2 = {}, a3 = {};
  for (int k0 = 0; k0 < E_DIM; k0 += 32) {
    v16bf a = load_a_bf16(X, mt, k0, E_DIM, lane);   // A reused 4x
    a0 = wmma_bf16(a, load_b_bf16(Wbf, jt, k0, E_DIM, lane), a0);
    a1 = wmma_bf16(a, load_b_bf16(Wbf, jt + 16, k0, E_DIM, lane), a1);
    a2 = wmma_bf16(a, load_b_bf16(Wbf, jt + 32, k0, E_DIM, lane), a2);
    a3 = wmma_bf16(a, load_b_bf16(Wbf, jt + 48, k0, E_DIM, lane), a3);
  }
  int qcol = lane & 15, hi = lane >> 4;
  v8f accs[4] = {a0, a1, a2, a3};
#pragma unroll
  for (int c = 0; c < 4; ++c) {
    int col = jt + c * 16 + qcol;       // within [0, 3E)
    float bv = ipb[col];
    int colE = col & (E_DIM - 1);
    int h = colE / DH, d = colE % DH;
#pragma unroll
    for (int r = 0; r < 8; ++r) {
      int m = mt + r + 8 * hi;          // flat (l,n) row: m = l*N + n
      int l = m / N_B, n = m % N_B;
      float v = accs[c][r] + bv;
      size_t nh = (size_t)(n * H_N + h);
      if (which == 0)
        Qo[(nh * L_SEQ + l) * DH + d] = (bf16_t)(v * 0.125f);  // fold 1/sqrt(DH)
      else if (which == 1)
        Ko[(nh * L_SEQ + l) * DH + d] = (bf16_t)v;
      else
        Vto[(nh * DH + d) * L_SEQ + l] = (bf16_t)v;            // V transposed
    }
  }
}

// ---- Kernel 2: flash attention, one wave per (n,h, 16-query tile) ----
// grid: (L/16, N*H), block 32.
__global__ __launch_bounds__(32) void attn_kernel(
    const bf16_t* __restrict__ Qs, const bf16_t* __restrict__ Ks,
    const bf16_t* __restrict__ Vt, const int* __restrict__ matched,
    const float* __restrict__ rpb, bf16_t* __restrict__ AO) {
  __shared__ __align__(32) bf16_t plds[512];  // P^T staged in B-operand order
  int lane = threadIdx.x;
  int nh = blockIdx.y;
  int n = nh / H_N, h = nh % H_N;
  int qb = blockIdx.x * 16;
  const bf16_t* Q = Qs + (size_t)nh * L_SEQ * DH;
  const bf16_t* K = Ks + (size_t)nh * L_SEQ * DH;
  const bf16_t* V = Vt + (size_t)nh * DH * L_SEQ;
  int qcol = lane & 15, hi = lane >> 4;
  int qg = qb + qcol;
  int mi = (qg < T_N) ? matched[n * T_N + qg] : -1;
  float bh = rpb[h];
  // Q as B operand: B[d][q] = Q[q][d]; held for the whole kv loop.
  v16bf bq0 = load_b_bf16(Q, qb, 0, DH, lane);
  v16bf bq1 = load_b_bf16(Q, qb, 32, DH, lane);
  v8f o0 = {}, o1 = {}, o2 = {}, o3 = {};  // O^T: 64 DH rows x 16 q cols
  float mrun = -3.0e38f, lrun = 0.f;
  for (int kv0 = 0; kv0 < L_SEQ; kv0 += 32) {
    // S^T = K_tile x Q^T : lane = q col, vgpr row = kv
    v8f s0 = {}, s1 = {};
    s0 = wmma_bf16(load_a_bf16(K, kv0, 0, DH, lane), bq0, s0);
    s0 = wmma_bf16(load_a_bf16(K, kv0, 32, DH, lane), bq1, s0);
    s1 = wmma_bf16(load_a_bf16(K, kv0 + 16, 0, DH, lane), bq0, s1);
    s1 = wmma_bf16(load_a_bf16(K, kv0 + 16, 32, DH, lane), bq1, s1);
    float tmax = -3.0e38f;
#pragma unroll
    for (int r = 0; r < 8; ++r) {
      int kvl = r + 8 * hi;
      if (kv0 + kvl == mi) s0[r] += bh;          // matched-index bias
      if (kv0 + 16 + kvl == mi) s1[r] += bh;
      tmax = fmaxf(tmax, fmaxf(s0[r], s1[r]));
    }
    tmax = fmaxf(tmax, __shfl_xor(tmax, 16));    // combine kv halves
    float mnew = fmaxf(mrun, tmax);
    float corr = __expf(mrun - mnew);
    float psum = 0.f;
#pragma unroll
    for (int r = 0; r < 8; ++r) {
      s0[r] = __expf(s0[r] - mnew);
      s1[r] = __expf(s1[r] - mnew);
      psum += s0[r] + s1[r];
    }
    psum += __shfl_xor(psum, 16);
    lrun = lrun * corr + psum;
    mrun = mnew;
#pragma unroll
    for (int r = 0; r < 8; ++r) {
      o0[r] *= corr; o1[r] *= corr; o2[r] *= corr; o3[r] *= corr;
    }
    // Stage P^T into LDS so it can be re-read in B-operand order:
    // element (kv,q) lives at lane (q + 16*(kv>=16)), slot kv&15.
    __syncthreads();
#pragma unroll
    for (int r = 0; r < 8; ++r) {
      plds[qcol * 16 + (r + 8 * hi)] = (bf16_t)s0[r];
      plds[(qcol + 16) * 16 + (r + 8 * hi)] = (bf16_t)s1[r];
    }
    __syncthreads();
    v16bf bp;
    {
      const v8bf* pl = (const v8bf*)&plds[lane * 16];
      v8bf lo = pl[0], hi8 = pl[1];
#pragma unroll
      for (int i = 0; i < 8; ++i) { bp[i] = lo[i]; bp[8 + i] = hi8[i]; }
    }
    // O^T[dchunk] += V^T_chunk(16xkv32) x P^T(kv32 x q16)
    o0 = wmma_bf16(load_a_bf16(V, 0,  kv0, L_SEQ, lane), bp, o0);
    o1 = wmma_bf16(load_a_bf16(V, 16, kv0, L_SEQ, lane), bp, o1);
    o2 = wmma_bf16(load_a_bf16(V, 32, kv0, L_SEQ, lane), bp, o2);
    o3 = wmma_bf16(load_a_bf16(V, 48, kv0, L_SEQ, lane), bp, o3);
  }
  float inv = 1.0f / lrun;
  size_t rowbase = ((size_t)qg * N_B + n) * E_DIM + (size_t)h * DH;
#pragma unroll
  for (int r = 0; r < 8; ++r) {
    AO[rowbase + 0  + r + 8 * hi] = (bf16_t)(o0[r] * inv);
    AO[rowbase + 16 + r + 8 * hi] = (bf16_t)(o1[r] * inv);
    AO[rowbase + 32 + r + 8 * hi] = (bf16_t)(o2[r] * inv);
    AO[rowbase + 48 + r + 8 * hi] = (bf16_t)(o3[r] * inv);
  }
}

// ---- Kernel 3: output projection, 16x64 tile per wave ----
// grid: ((L*N)/16, E/64), block 32.
__global__ __launch_bounds__(32) void out_proj_kernel(
    const bf16_t* __restrict__ AO, const bf16_t* __restrict__ Wbf,
    const float* __restrict__ bias, float* __restrict__ out) {
  int lane = threadIdx.x;
  int mt = blockIdx.x * 16, jt = blockIdx.y * 64;
  v8f a0 = {}, a1 = {}, a2 = {}, a3 = {};
  for (int k0 = 0; k0 < E_DIM; k0 += 32) {
    v16bf a = load_a_bf16(AO, mt, k0, E_DIM, lane);
    a0 = wmma_bf16(a, load_b_bf16(Wbf, jt, k0, E_DIM, lane), a0);
    a1 = wmma_bf16(a, load_b_bf16(Wbf, jt + 16, k0, E_DIM, lane), a1);
    a2 = wmma_bf16(a, load_b_bf16(Wbf, jt + 32, k0, E_DIM, lane), a2);
    a3 = wmma_bf16(a, load_b_bf16(Wbf, jt + 48, k0, E_DIM, lane), a3);
  }
  int qcol = lane & 15, hi = lane >> 4;
  v8f accs[4] = {a0, a1, a2, a3};
#pragma unroll
  for (int c = 0; c < 4; ++c) {
    int col = jt + c * 16 + qcol;
    float bv = bias[col];
#pragma unroll
    for (int r = 0; r < 8; ++r)
      out[(size_t)(mt + r + 8 * hi) * E_DIM + col] = accs[c][r] + bv;
  }
}

extern "C" void kernel_launch(void* const* d_in, const int* in_sizes, int n_in,
                              void* d_out, int out_size, void* d_ws,
                              size_t ws_size, hipStream_t stream) {
  (void)in_sizes; (void)n_in; (void)out_size; (void)ws_size;
  const float* query = (const float*)d_in[0];
  const float* key   = (const float*)d_in[1];
  const float* value = (const float*)d_in[2];
  const int*   matched = (const int*)d_in[3];
  const float* ipw = (const float*)d_in[4];
  const float* ipb = (const float*)d_in[5];
  const float* opw = (const float*)d_in[6];
  const float* opb = (const float*)d_in[7];
  const float* rpb = (const float*)d_in[8];
  float* out = (float*)d_out;

  const size_t xel = (size_t)L_SEQ * N_B * E_DIM;   // 4 Mi elems
  const size_t qkv_elems = (size_t)N_B * H_N * L_SEQ * DH;  // == xel
  bf16_t* p = (bf16_t*)d_ws;
  bf16_t* Qs  = p; p += qkv_elems;
  bf16_t* Ks  = p; p += qkv_elems;
  bf16_t* Vt  = p; p += qkv_elems;
  bf16_t* AO  = p; p += xel;
  bf16_t* Xq  = p; p += xel;
  bf16_t* Xk  = p; p += xel;
  bf16_t* Xv  = p; p += xel;
  bf16_t* Wip = p; p += (size_t)3 * E_DIM * E_DIM;
  bf16_t* Wop = p; p += (size_t)E_DIM * E_DIM;

  // One-shot conversions to bf16 (HBM-bound, trivial vs GEMM work).
  int n8x = (int)(xel / 8);
  cvt_bf16_kernel<<<(n8x + 255) / 256, 256, 0, stream>>>(query, Xq, n8x);
  cvt_bf16_kernel<<<(n8x + 255) / 256, 256, 0, stream>>>(key, Xk, n8x);
  cvt_bf16_kernel<<<(n8x + 255) / 256, 256, 0, stream>>>(value, Xv, n8x);
  int n8w = 3 * E_DIM * E_DIM / 8;
  cvt_bf16_kernel<<<(n8w + 255) / 256, 256, 0, stream>>>(ipw, Wip, n8w);
  int n8o = E_DIM * E_DIM / 8;
  cvt_bf16_kernel<<<(n8o + 255) / 256, 256, 0, stream>>>(opw, Wop, n8o);

  dim3 g1((L_SEQ * N_B) / 16, (3 * E_DIM) / 64);
  qkv_proj_kernel<<<g1, dim3(32), 0, stream>>>(Xq, Xk, Xv, Wip, ipb, Qs, Ks, Vt);

  dim3 g2(L_SEQ / 16, N_B * H_N);
  attn_kernel<<<g2, dim3(32), 0, stream>>>(Qs, Ks, Vt, matched, rpb, AO);

  dim3 g3((L_SEQ * N_B) / 16, E_DIM / 64);
  out_proj_kernel<<<g3, dim3(32), 0, stream>>>(AO, Wop, opb, out);
}
